// TensorMemory_11218454577697
// MI455X (gfx1250) — compile-verified, matrix operand order checked by reading, not run on hardware
//
#include <hip/hip_runtime.h>

// TensorMemory (linear attention memory update + retrieve) for MI455X / gfx1250.
//
// Precision plan (roofline): the two GEMMs total ~34 GFLOP; f32 WMMA (K=4) would
// make this compute-bound, bf16 WMMA (K=32, f32 accum) makes it ~memory-bound at
// 23.3 TB/s (~256MB compulsory traffic). So: bf16 operands, f32 accumulators.
// sigma_k / values are pre-transposed to [D][K] bf16 so every WMMA fragment is
// two contiguous 16B loads per lane; the 512x512 operands of GEMM2 (M^T) and the
// 32MB bf16 operands of GEMM1 live in the 192MB L2 across tile reuse.

typedef __attribute__((ext_vector_type(16))) __bf16 bf16x16;
typedef __attribute__((ext_vector_type(8)))  __bf16 bf16x8;
typedef __attribute__((ext_vector_type(8)))  float  f32x8;
typedef __attribute__((ext_vector_type(4)))  float  f32x4;

union Frag { bf16x16 v; bf16x8 h[2]; };
union AccU { f32x8 v; f32x4 q[2]; };

__device__ __forceinline__ float elu1(float x) {
  // elu(x)+1 = x+1 (x>0) else exp(x)
  return x > 0.f ? x + 1.f : __expf(x);
}

static constexpr int BB = 8, S = 4096, D = 512;
static constexpr int K  = BB * S;          // 32768 reduction length of GEMM1
static constexpr int NKB = K / 64;         // 512 prep tiles along k
static constexpr int NSLICE = 8;           // split-K factor for GEMM1
static constexpr int KSL = K / NSLICE;     // 4096 k per slice
static constexpr float EPS = 1e-6f;

// ---------------------------------------------------------------------------
// Kernel 1: tiled transpose + elu + f32->bf16 of keys/values, plus per-block
// column partial sums of sigma_k (for z). 64x64 tiles staged through LDS.
// ---------------------------------------------------------------------------
__global__ __launch_bounds__(256)
void prep_kernel(const float* __restrict__ keys, const float* __restrict__ values,
                 __bf16* __restrict__ skT, __bf16* __restrict__ vT,
                 float* __restrict__ zpart) {
  __shared__ __bf16 Ts[64][72];   // padded to dodge bank conflicts
  __shared__ __bf16 Tv[64][72];
  __shared__ float  zs[256];

  const int t  = threadIdx.x;
  const int kb = blockIdx.x, db = blockIdx.y;
  const int k0 = kb * 64, d0 = db * 64;
  const int dl = t & 63, kq = t >> 6;      // kq in 0..3

  float zsum = 0.f;
#pragma unroll
  for (int i = 0; i < 16; ++i) {
    const int kl = kq + i * 4;
    const size_t idx = (size_t)(k0 + kl) * D + (d0 + dl);  // coalesced along d
    const float sk = elu1(keys[idx]);
    Ts[kl][dl] = (__bf16)sk;
    Tv[kl][dl] = (__bf16)values[idx];
    zsum += sk;
  }
  zs[t] = zsum;
  __syncthreads();

  if (t < 64) {   // reduce 4 partials per d-column, write deterministic partials
    zpart[(size_t)kb * D + d0 + t] = zs[t] + zs[t + 64] + zs[t + 128] + zs[t + 192];
  }

  // transposed writes: consecutive threads -> consecutive k (coalesced bf16)
#pragma unroll
  for (int i = 0; i < 16; ++i) {
    const int o = t + i * 256;
    const int dl2 = o >> 6, kl2 = o & 63;
    const size_t dst = (size_t)(d0 + dl2) * K + (k0 + kl2);
    skT[dst] = Ts[kl2][dl2];
    vT[dst]  = Tv[kl2][dl2];
  }
}

// ---------------------------------------------------------------------------
// Kernel 2: z_new[d] = z[d] + sum_k sigma_k[k,d] / B   (fixed-order reduce)
// ---------------------------------------------------------------------------
__global__ __launch_bounds__(512)
void zfin_kernel(const float* __restrict__ z_in, const float* __restrict__ zpart,
                 float* __restrict__ znew) {
  const int d = threadIdx.x;
  float s = 0.f;
  for (int kb = 0; kb < NKB; ++kb) s += zpart[(size_t)kb * D + d];
  znew[d] = z_in[d] + s * (1.f / (float)BB);
}

// ---------------------------------------------------------------------------
// Kernel 3: GEMM1 (split-K):  delta[d][e] = sum_k skT[d][k] * vT[e][k]
// One wave per 64x64 output tile per k-slice; 16 x v_wmma_f32_16x16x32_bf16
// per 32-wide K-step. Partials written transposed as parts[slice][e][d].
// ---------------------------------------------------------------------------
__global__ __launch_bounds__(32)
void gemm1_kernel(const __bf16* __restrict__ skT, const __bf16* __restrict__ vT,
                  float* __restrict__ parts) {
  const int lane = threadIdx.x;
  const int row  = lane & 15, hi = lane >> 4;          // hi selects K sub-group
  const int tile = blockIdx.x;                          // 0..63
  const int dBlk = (tile >> 3) * 64, eBlk = (tile & 7) * 64;
  const int slice = blockIdx.y;
  const size_t kbase0 = (size_t)slice * KSL;

  f32x8 acc[4][4] = {};

  for (int kk = 0; kk < KSL; kk += 32) {
    const size_t g0 = kbase0 + kk + hi * 8;             // K groups per ISA layout
    const size_t g1 = g0 + 16;

    Frag A[4], Bf[4];
#pragma unroll
    for (int i = 0; i < 4; ++i) {
      const size_t r = (size_t)(dBlk + i * 16 + row) * K;
      A[i].h[0] = *(const bf16x8*)(skT + r + g0);
      A[i].h[1] = *(const bf16x8*)(skT + r + g1);
      __builtin_prefetch(skT + r + g0 + 32, 0, 1);      // global_prefetch_b8
    }
#pragma unroll
    for (int j = 0; j < 4; ++j) {
      const size_t r = (size_t)(eBlk + j * 16 + row) * K;
      Bf[j].h[0] = *(const bf16x8*)(vT + r + g0);
      Bf[j].h[1] = *(const bf16x8*)(vT + r + g1);
      __builtin_prefetch(vT + r + g0 + 32, 0, 1);
    }
#pragma unroll
    for (int i = 0; i < 4; ++i)
#pragma unroll
      for (int j = 0; j < 4; ++j)
        acc[i][j] = __builtin_amdgcn_wmma_f32_16x16x32_bf16(
            false, A[i].v, false, Bf[j].v, (short)0, acc[i][j], false, false);
  }

  // Transposed store: C-layout gives each lane 8 consecutive d per (i,j) -> 2x b128
#pragma unroll
  for (int i = 0; i < 4; ++i)
#pragma unroll
    for (int j = 0; j < 4; ++j) {
      AccU u; u.v = acc[i][j];
      const int e = eBlk + j * 16 + row;
      const size_t base = ((size_t)slice * D + e) * D + (dBlk + i * 16 + hi * 8);
      *(f32x4*)(parts + base)     = u.q[0];
      *(f32x4*)(parts + base + 4) = u.q[1];
    }
}

// ---------------------------------------------------------------------------
// Kernel 4: MT[e][d] = bf16( M[d][e] + sum_slices parts[sl][e][d] / (B*S) )
// ---------------------------------------------------------------------------
__global__ __launch_bounds__(256)
void mtfin_kernel(const float* __restrict__ M, const float* __restrict__ parts,
                  __bf16* __restrict__ MT) {
  const int idx = blockIdx.x * 256 + threadIdx.x;       // = e*512 + d
  const int e = idx >> 9, d = idx & 511;
  float s = 0.f;
#pragma unroll
  for (int sl = 0; sl < NSLICE; ++sl)
    s += parts[((size_t)sl * D + e) * D + d];
  MT[idx] = (__bf16)(M[(size_t)d * D + e] + s * (1.f / (float)K));
}

// ---------------------------------------------------------------------------
// Kernel 5: GEMM2 + fused norm + divide:
//   out[s][e] = (sum_d sq[s][d] * M_new[d][e]) / (sum_d sq[s][d]*z_new[d] + eps)
// sq = elu(q)+1 computed on the fly (rows of q are already K-contiguous).
// ---------------------------------------------------------------------------
__global__ __launch_bounds__(32)
void gemm2_kernel(const float* __restrict__ Q, const __bf16* __restrict__ MT,
                  const float* __restrict__ znew, float* __restrict__ out) {
  const int lane = threadIdx.x;
  const int row  = lane & 15, hi = lane >> 4;
  const int s0 = blockIdx.x * 64;
  const int e0 = blockIdx.y * 64;

  f32x8 acc[4][4] = {};
  float normp[4] = {0.f, 0.f, 0.f, 0.f};

  for (int d0 = 0; d0 < D; d0 += 32) {
    const int g0 = d0 + hi * 8, g1 = g0 + 16;
    const f32x4 z0 = *(const f32x4*)(znew + g0);
    const f32x4 z1 = *(const f32x4*)(znew + g0 + 4);
    const f32x4 z2 = *(const f32x4*)(znew + g1);
    const f32x4 z3 = *(const f32x4*)(znew + g1 + 4);

    Frag A[4], Bf[4];
#pragma unroll
    for (int i = 0; i < 4; ++i) {
      const size_t r = (size_t)(s0 + i * 16 + row) * D;
      const f32x4 q0 = *(const f32x4*)(Q + r + g0);
      const f32x4 q1 = *(const f32x4*)(Q + r + g0 + 4);
      const f32x4 q2 = *(const f32x4*)(Q + r + g1);
      const f32x4 q3 = *(const f32x4*)(Q + r + g1 + 4);
      float n = normp[i];
#pragma unroll
      for (int c = 0; c < 4; ++c) {
        const float a0 = elu1(q0[c]); A[i].h[0][c]     = (__bf16)a0; n += a0 * z0[c];
        const float a1 = elu1(q1[c]); A[i].h[0][c + 4] = (__bf16)a1; n += a1 * z1[c];
        const float a2 = elu1(q2[c]); A[i].h[1][c]     = (__bf16)a2; n += a2 * z2[c];
        const float a3 = elu1(q3[c]); A[i].h[1][c + 4] = (__bf16)a3; n += a3 * z3[c];
      }
      normp[i] = n;
    }
#pragma unroll
    for (int j = 0; j < 4; ++j) {
      const size_t r = (size_t)(e0 + j * 16 + row) * D;   // MT is [e][d]
      Bf[j].h[0] = *(const bf16x8*)(MT + r + g0);
      Bf[j].h[1] = *(const bf16x8*)(MT + r + g1);
    }
#pragma unroll
    for (int i = 0; i < 4; ++i)
#pragma unroll
      for (int j = 0; j < 4; ++j)
        acc[i][j] = __builtin_amdgcn_wmma_f32_16x16x32_bf16(
            false, A[i].v, false, Bf[j].v, (short)0, acc[i][j], false, false);
  }

  // finish norm: lanes L and L^16 hold complementary K halves of the same row
  float nf[4];
#pragma unroll
  for (int i = 0; i < 4; ++i)
    nf[i] = normp[i] + __shfl_xor(normp[i], 16, 32);

  // epilogue: divide by (norm+eps) using cross-lane broadcast of row norms
#pragma unroll
  for (int i = 0; i < 4; ++i) {
#pragma unroll
    for (int qi = 0; qi < 8; ++qi) {
      const int rloc = hi * 8 + qi;                 // local row of this element
      const float nv = __shfl(nf[i], rloc, 32);     // norm held by lane==rloc
      const float inv = 1.f / (nv + EPS);
      const int srow = s0 + i * 16 + rloc;
#pragma unroll
      for (int j = 0; j < 4; ++j)
        out[(size_t)srow * D + (e0 + j * 16 + row)] = acc[i][j][qi] * inv;
    }
  }
}

// ---------------------------------------------------------------------------
extern "C" void kernel_launch(void* const* d_in, const int* in_sizes, int n_in,
                              void* d_out, int out_size, void* d_ws, size_t ws_size,
                              hipStream_t stream) {
  const float* keys    = (const float*)d_in[0];
  const float* values  = (const float*)d_in[1];
  const float* queries = (const float*)d_in[2];
  const float* M       = (const float*)d_in[3];
  const float* z       = (const float*)d_in[4];
  float* out = (float*)d_out;

  char* ws = (char*)d_ws;
  __bf16* skT  = (__bf16*)(ws);                              // 32 MB  [D][K] bf16
  __bf16* vT   = (__bf16*)(ws + (size_t)32 * (1u << 20));    // 32 MB  [D][K] bf16
  float*  zpart= (float*) (ws + (size_t)64 * (1u << 20));    //  1 MB  [NKB][D]
  float*  znew = (float*) (ws + (size_t)65 * (1u << 20));    //  2 KB  [D]
  float*  parts= (float*) (ws + (size_t)66 * (1u << 20));    //  8 MB  [8][D][D]
  __bf16* MT   = (__bf16*)(ws + (size_t)74 * (1u << 20));    // .5 MB  [D][D] bf16

  prep_kernel <<<dim3(NKB, D / 64), 256, 0, stream>>>(keys, values, skT, vT, zpart);
  zfin_kernel <<<1, D, 0, stream>>>(z, zpart, znew);
  gemm1_kernel<<<dim3(64, NSLICE), 32, 0, stream>>>(skT, vT, parts);
  mtfin_kernel<<<(D * D) / 256, 256, 0, stream>>>(M, parts, MT);
  gemm2_kernel<<<dim3(K / 64, D / 64), 32, 0, stream>>>(queries, MT, znew, out);
}
